// MultiHeadDotProduct_31619549233291
// MI455X (gfx1250) — compile-verified, matrix operand order checked by reading, not run on hardware
//
#include <hip/hip_runtime.h>
#include <hip/hip_bf16.h>

#define N_NODES 8192
#define DMODEL  256
#define NHEADS  4
#define DHEAD   64
#define NEDGES  262144
#define EDGE_DIM 16

// GEMM tiling
#define BM 64
#define BN 128
#define BK 32
#define ROWB 80          // padded LDS row stride in bytes (64B data + 16B pad)
#define LDS_A_BYTES (BM * ROWB)      // 5120
#define LDS_B_BYTES (BN * ROWB)      // 10240
#define LDS_BUF_BYTES (LDS_A_BYTES + LDS_B_BYTES)

typedef __attribute__((ext_vector_type(16))) __bf16 v16bf;
typedef __attribute__((ext_vector_type(8)))  float  v8f;

// ---------------------------------------------------------------------------
// fp32 -> bf16 elementwise conversion
// ---------------------------------------------------------------------------
__global__ __launch_bounds__(256)
void f32_to_bf16_kernel(const float* __restrict__ src, __bf16* __restrict__ dst, int n) {
    int i = blockIdx.x * blockDim.x + threadIdx.x;
    if (i < n) dst[i] = (__bf16)src[i];
}

// ---------------------------------------------------------------------------
// CDNA5 async global->LDS copy of 16 bytes per lane (ASYNCcnt-tracked).
// lds_off: byte offset in LDS; gptr: global address.
// ---------------------------------------------------------------------------
__device__ __forceinline__ void async_copy_b128(unsigned int lds_off, const void* gptr) {
    asm volatile("global_load_async_to_lds_b128 %0, %1, off"
                 :: "v"(lds_off), "v"(gptr)
                 : "memory");
}

__device__ __forceinline__ void wait_asynccnt_le3() {
    asm volatile("s_wait_asynccnt 0x3" ::: "memory");
}
__device__ __forceinline__ void wait_asynccnt_0() {
    asm volatile("s_wait_asynccnt 0x0" ::: "memory");
}

// ---------------------------------------------------------------------------
// Read one 16x32 bf16 WMMA operand fragment from an LDS tile.
// Tile rows are ROWB bytes apart; row holds 32 bf16 (64B) + 16B pad.
// lane 0-15: row = r16, elems {0..7, 16..23} of the 32-wide K slice
// lane 16-31: row = r16, elems {8..15, 24..31}   (half16 pre-applied by caller)
// ---------------------------------------------------------------------------
__device__ __forceinline__ v16bf frag_from_lds(const char* tile, int row, int half16) {
    const char* p = tile + row * ROWB + half16 * 16;
    union U { uint4 u; __bf16 h[8]; } a, b;
    a.u = *reinterpret_cast<const uint4*>(p);        // K +0..+7  (rel. to half)
    b.u = *reinterpret_cast<const uint4*>(p + 32);   // K +16..+23
    v16bf f;
#pragma unroll
    for (int i = 0; i < 8; ++i) { f[i] = a.h[i]; f[i + 8] = b.h[i]; }
    return f;
}

// ---------------------------------------------------------------------------
// Out[m][n] = sum_k A[m][k]*W[n][k] + bias[n]
// A: Mx256 bf16 row-major, W: 256x256 bf16 row-major (acts as B^T).
// grid = (M/BM, 256/BN), block = 256 threads = 8 waves (2 M-waves x 4 N-waves).
// Double-buffered async global->LDS staging; 32 v_wmma per wave.
// ---------------------------------------------------------------------------
__global__ __launch_bounds__(256)
void gemm_bf16_wmma_kernel(const __bf16* __restrict__ A,
                           const __bf16* __restrict__ W,
                           const float*  __restrict__ bias,
                           float*        __restrict__ Out) {
    __shared__ char smem[2 * LDS_BUF_BYTES];

    const int tid  = threadIdx.x;
    const int wave = tid >> 5;
    const int lane = tid & 31;
    const int half16 = lane >> 4;
    const int l16 = lane & 15;
    const int waveM = wave >> 2;          // 0..1
    const int waveN = wave & 3;           // 0..3
    const int m0 = blockIdx.x * BM;
    const int n0 = blockIdx.y * BN;

    const unsigned int smemBase = (unsigned int)(size_t)(&smem[0]);

    // staging assignment: A tile = 256 chunks of 16B (1/thread),
    //                     B tile = 512 chunks of 16B (2/thread)
    const int cr = tid >> 2;              // 0..63
    const int cc = tid & 3;               // 0..3  (16B chunk within 64B row)

    v8f acc00 = {}, acc01 = {}, acc10 = {}, acc11 = {};

    // ---- prologue: stage k-step 0 into buffer 0 ----
    {
        const char* ga  = (const char*)(A + (size_t)(m0 + cr) * DMODEL) + cc * 16;
        const char* gb0 = (const char*)(W + (size_t)(n0 + cr) * DMODEL) + cc * 16;
        const char* gb1 = (const char*)(W + (size_t)(n0 + 64 + cr) * DMODEL) + cc * 16;
        async_copy_b128(smemBase + cr * ROWB + cc * 16, ga);
        async_copy_b128(smemBase + LDS_A_BYTES + cr * ROWB + cc * 16, gb0);
        async_copy_b128(smemBase + LDS_A_BYTES + (64 + cr) * ROWB + cc * 16, gb1);
    }

    int cur = 0;
#pragma unroll
    for (int s = 0; s < DMODEL / BK; ++s) {
        // ---- prefetch next k-step into the other buffer ----
        if (s < DMODEL / BK - 1) {
            const int kn = (s + 1) * BK;
            const unsigned int nb = smemBase + (cur ^ 1) * LDS_BUF_BYTES;
            const char* ga  = (const char*)(A + (size_t)(m0 + cr) * DMODEL + kn) + cc * 16;
            const char* gb0 = (const char*)(W + (size_t)(n0 + cr) * DMODEL + kn) + cc * 16;
            const char* gb1 = (const char*)(W + (size_t)(n0 + 64 + cr) * DMODEL + kn) + cc * 16;
            async_copy_b128(nb + cr * ROWB + cc * 16, ga);
            async_copy_b128(nb + LDS_A_BYTES + cr * ROWB + cc * 16, gb0);
            async_copy_b128(nb + LDS_A_BYTES + (64 + cr) * ROWB + cc * 16, gb1);
            wait_asynccnt_le3();          // in-order: current step's 3 copies landed
        } else {
            wait_asynccnt_0();
        }
        __syncthreads();                  // all waves' chunks visible

        // ---- compute from buffer `cur` ----
        const char* At = smem + cur * LDS_BUF_BYTES;
        const char* Bt = At + LDS_A_BYTES;
        v16bf a0 = frag_from_lds(At, waveM * 32 +  0 + l16, half16);
        v16bf a1 = frag_from_lds(At, waveM * 32 + 16 + l16, half16);
        v16bf b0 = frag_from_lds(Bt, waveN * 32 +  0 + l16, half16);
        v16bf b1 = frag_from_lds(Bt, waveN * 32 + 16 + l16, half16);
        acc00 = __builtin_amdgcn_wmma_f32_16x16x32_bf16(false, a0, false, b0, (short)0, acc00, false, false);
        acc01 = __builtin_amdgcn_wmma_f32_16x16x32_bf16(false, a0, false, b1, (short)0, acc01, false, false);
        acc10 = __builtin_amdgcn_wmma_f32_16x16x32_bf16(false, a1, false, b0, (short)0, acc10, false, false);
        acc11 = __builtin_amdgcn_wmma_f32_16x16x32_bf16(false, a1, false, b1, (short)0, acc11, false, false);

        __syncthreads();                  // done reading before buffer is re-staged
        cur ^= 1;
    }

    // ---- epilogue: C/D layout -> global ----
    // VGPR r, lanes 0-15: (M = mt + r, N = nt + l16); lanes 16-31: (M = mt + 8 + r)
    const int mwb = m0 + waveM * 32 + half16 * 8;
    const int nwb = n0 + waveN * 32 + l16;
#pragma unroll
    for (int tm = 0; tm < 2; ++tm) {
#pragma unroll
        for (int tn = 0; tn < 2; ++tn) {
            const v8f* acc = (tm == 0) ? ((tn == 0) ? &acc00 : &acc01)
                                       : ((tn == 0) ? &acc10 : &acc11);
            const int n = nwb + tn * 16;
            const float bv = bias[n];
            const int mb = mwb + tm * 16;
#pragma unroll
            for (int r = 0; r < 8; ++r)
                Out[(size_t)(mb + r) * DMODEL + n] = (*acc)[r] + bv;
        }
    }
}

// ---------------------------------------------------------------------------
// Per (edge, head): s = (q[row,h*64:...] . k[col,...]) / 8 ; es = exp(s)
// expsc[e*4+h] = es ; denom[h*N+row] += es
// ---------------------------------------------------------------------------
__global__ __launch_bounds__(256)
void edge_score_kernel(const int* __restrict__ ei,
                       const float* __restrict__ q,
                       const float* __restrict__ k,
                       float* __restrict__ expsc,
                       float* __restrict__ denom) {
    int idx = blockIdx.x * blockDim.x + threadIdx.x;   // < E * H
    int e = idx >> 2, h = idx & 3;
    int row = ei[2 * e + 0];
    int col = ei[2 * e + 1];
    const float4* qp = reinterpret_cast<const float4*>(q + (size_t)row * DMODEL + h * DHEAD);
    const float4* kp = reinterpret_cast<const float4*>(k + (size_t)col * DMODEL + h * DHEAD);
    float s = 0.f;
#pragma unroll
    for (int i = 0; i < DHEAD / 4; ++i) {
        float4 a = qp[i], b = kp[i];
        s += a.x * b.x + a.y * b.y + a.z * b.z + a.w * b.w;
    }
    float es = __expf(s * 0.125f);   // 1/sqrt(64)
    expsc[idx] = es;
    unsafeAtomicAdd(&denom[h * N_NODES + row], es);
}

// ---------------------------------------------------------------------------
// agg[row, d:d+4] += (expsc[e,h]/denom[h,row]) * v[col, d:d+4]
// one thread per (edge, 4-float chunk): E*64 threads
// ---------------------------------------------------------------------------
__global__ __launch_bounds__(256)
void aggregate_kernel(const int* __restrict__ ei,
                      const float* __restrict__ v,
                      const float* __restrict__ expsc,
                      const float* __restrict__ denom,
                      float* __restrict__ agg) {
    int idx = blockIdx.x * blockDim.x + threadIdx.x;   // < E * 64
    int e = idx >> 6;
    int d = (idx & 63) * 4;
    int h = d >> 6;   // head = d / 64
    int row = ei[2 * e + 0];
    int col = ei[2 * e + 1];
    float w = expsc[e * 4 + h] / denom[h * N_NODES + row];
    float4 vv = *reinterpret_cast<const float4*>(v + (size_t)col * DMODEL + d);
    float* ap = agg + (size_t)row * DMODEL + d;
    unsafeAtomicAdd(ap + 0, w * vv.x);
    unsafeAtomicAdd(ap + 1, w * vv.y);
    unsafeAtomicAdd(ap + 2, w * vv.z);
    unsafeAtomicAdd(ap + 3, w * vv.w);
}

// ---------------------------------------------------------------------------
// host launcher
// ---------------------------------------------------------------------------
extern "C" void kernel_launch(void* const* d_in, const int* in_sizes, int n_in,
                              void* d_out, int out_size, void* d_ws, size_t ws_size,
                              hipStream_t stream) {
    const float* feats = (const float*)d_in[0];
    const int*   ei    = (const int*)d_in[1];
    const float* Wq = (const float*)d_in[3];  const float* bq = (const float*)d_in[4];
    const float* Wk = (const float*)d_in[5];  const float* bk = (const float*)d_in[6];
    const float* Wv = (const float*)d_in[7];  const float* bv = (const float*)d_in[8];
    const float* Wo = (const float*)d_in[9];  const float* bo = (const float*)d_in[10];
    float* out = (float*)d_out;

    // workspace layout (bytes)
    const size_t SZ_NM    = (size_t)N_NODES * DMODEL * sizeof(float);   // 8 MiB
    const size_t SZ_NM_BF = (size_t)N_NODES * DMODEL * sizeof(__bf16);  // 4 MiB
    const size_t SZ_W_BF  = (size_t)DMODEL * DMODEL * sizeof(__bf16);   // 128 KiB
    char* ws = (char*)d_ws;
    float*  qbuf   = (float*) (ws);
    float*  kbuf   = (float*) (ws + SZ_NM);
    float*  vbuf   = (float*) (ws + 2 * SZ_NM);
    float*  agg    = (float*) (ws + 3 * SZ_NM);
    __bf16* xbf    = (__bf16*)(ws + 4 * SZ_NM);
    __bf16* aggbf  = (__bf16*)(ws + 4 * SZ_NM + SZ_NM_BF);
    __bf16* wqbf   = (__bf16*)(ws + 4 * SZ_NM + 2 * SZ_NM_BF);
    __bf16* wkbf   = wqbf + (size_t)DMODEL * DMODEL;
    __bf16* wvbf   = wkbf + (size_t)DMODEL * DMODEL;
    __bf16* wobf   = wvbf + (size_t)DMODEL * DMODEL;
    float*  expsc  = (float*) (ws + 4 * SZ_NM + 2 * SZ_NM_BF + 4 * SZ_W_BF);
    float*  denom  = (float*) ((char*)expsc + (size_t)NEDGES * NHEADS * sizeof(float));

    // zero accumulation buffers (fresh every call; graph-capture safe)
    hipMemsetAsync(agg,   0, SZ_NM, stream);
    hipMemsetAsync(denom, 0, (size_t)NHEADS * N_NODES * sizeof(float), stream);

    // fp32 -> bf16 conversions
    const int nFeat = N_NODES * DMODEL;
    const int nW    = DMODEL * DMODEL;
    f32_to_bf16_kernel<<<(nFeat + 255) / 256, 256, 0, stream>>>(feats, xbf, nFeat);
    f32_to_bf16_kernel<<<(nW + 255) / 256, 256, 0, stream>>>(Wq, wqbf, nW);
    f32_to_bf16_kernel<<<(nW + 255) / 256, 256, 0, stream>>>(Wk, wkbf, nW);
    f32_to_bf16_kernel<<<(nW + 255) / 256, 256, 0, stream>>>(Wv, wvbf, nW);
    f32_to_bf16_kernel<<<(nW + 255) / 256, 256, 0, stream>>>(Wo, wobf, nW);

    // Q/K/V projections (WMMA, async-LDS double-buffered)
    dim3 ggrid(N_NODES / BM, DMODEL / BN);
    gemm_bf16_wmma_kernel<<<ggrid, 256, 0, stream>>>(xbf, wqbf, bq, qbuf);
    gemm_bf16_wmma_kernel<<<ggrid, 256, 0, stream>>>(xbf, wkbf, bk, kbuf);
    gemm_bf16_wmma_kernel<<<ggrid, 256, 0, stream>>>(xbf, wvbf, bv, vbuf);

    // sparse attention: scores + softmax denominators
    edge_score_kernel<<<(NEDGES * NHEADS) / 256, 256, 0, stream>>>(ei, qbuf, kbuf, expsc, denom);

    // weighted message aggregation
    aggregate_kernel<<<(NEDGES * 64) / 256, 256, 0, stream>>>(ei, vbuf, expsc, denom, agg);

    // output projection (WMMA) -> d_out
    f32_to_bf16_kernel<<<(nFeat + 255) / 256, 256, 0, stream>>>(agg, aggbf, nFeat);
    gemm_bf16_wmma_kernel<<<ggrid, 256, 0, stream>>>(aggbf, wobf, bo, out);

    // pass-through outputs: edge_index, edge_attr appended after `out`
    size_t tail = SZ_NM;
    hipMemcpyAsync((char*)d_out + tail, d_in[1],
                   (size_t)NEDGES * 2 * sizeof(int), hipMemcpyDeviceToDevice, stream);
    tail += (size_t)NEDGES * 2 * sizeof(int);
    hipMemcpyAsync((char*)d_out + tail, d_in[2],
                   (size_t)NEDGES * EDGE_DIM * sizeof(float), hipMemcpyDeviceToDevice, stream);
}
